// SwinBlock_51110110822668
// MI455X (gfx1250) — matrix operand order repaired and checked by reading, not compiled, not run on hardware
//
#include <hip/hip_runtime.h>
#include <hip/hip_bf16.h>

// ---------------------------------------------------------------------------
// Swin block, fused per-window kernel for gfx1250 (MI455X), wave32 + WMMA f16.
//   grid = 2048 windows (one workgroup per 64-token window), block = 256 (8 waves)
//   - all GEMMs via v_wmma_f32_16x16x32_f16, fragments in contiguous k order (k = 16*half + s)
//   - x window staged into LDS with global_load_async_to_lds_b128 (ASYNCcnt)
//   - software-pipelined B/A fragment loads in every GEMM inner loop
//   - coalesced b128 stores for attention output; fc2 output staged in LDS
//     so the dominant 2x201MB output+residual HBM traffic is 32B-contiguous
//   Workspace: packed f16 weight fragments (~2.4 MB) + expanded rel-bias
//   (192 KB) + attention-output bounce buffer (96 MB). Assumes ws >= 104 MB.
// ---------------------------------------------------------------------------

typedef __attribute__((ext_vector_type(16))) _Float16 v16h;
typedef __attribute__((ext_vector_type(8)))  float    v8f;
typedef __attribute__((ext_vector_type(4)))  unsigned u32x4;

union F16x16 { v16h v; u32x4 q[2]; };

__device__ __forceinline__ v8f wmma_f16(v16h a, v16h b, v8f c) {
  return __builtin_amdgcn_wmma_f32_16x16x32_f16(false, a, false, b, (short)0, c, false, false);
}

// A-fragment (16x32 f16, MxK): lane m=lane&15, half=lane>>4.
// element s -> K = half*16 + s (contiguous)
__device__ __forceinline__ v16h load_afrag(const _Float16* A, int row0, int lda, int col0, int lane) {
  int m = lane & 15, hf = (lane >> 4) & 1;
  const _Float16* s = A + (size_t)(row0 + m) * lda + col0 + hf * 16;
  F16x16 f;
  f.q[0] = *(const u32x4*)s;
  f.q[1] = *(const u32x4*)(s + 8);
  return f.v;
}

// B-fragment (32x16 f16, KxN): lane n=lane&15, half=lane>>4 holds K=half*16..+15
// of column n: 16 consecutive halfs when source is [n][k] row-major.
__device__ __forceinline__ v16h load_bfrag(const _Float16* s) {
  F16x16 f;
  f.q[0] = *(const u32x4*)s;
  f.q[1] = *(const u32x4*)(s + 8);
  return f.v;
}

// Pre-swizzled weight fragment: [tile][lane][8 x u32], 32B contiguous per lane.
__device__ __forceinline__ v16h load_bpack(const unsigned* base, int tile, int lane) {
  const u32x4* p = (const u32x4*)(base + ((size_t)tile * 32 + lane) * 8);
  F16x16 f;
  f.q[0] = p[0];
  f.q[1] = p[1];
  return f.v;
}

// Software-pipelined 16x(2x16) output pair over K: prefetch kt+1 fragments
// before issuing kt's WMMAs so global/LDS latency overlaps the XDL pipe.
template <int KT, int NTILES>
__device__ __forceinline__ void gemm_pair(const _Float16* A, int lda, int mt,
                                          const unsigned* P, int nt, int lane,
                                          v8f& a0, v8f& a1) {
  v16h af = load_afrag(A, mt * 16, lda, 0, lane);
  v16h b0 = load_bpack(P, nt, lane);
  v16h b1 = load_bpack(P, nt + 1, lane);
#pragma unroll
  for (int kt = 0; kt < KT; ++kt) {
    int ktn = (kt + 1 < KT) ? kt + 1 : kt;
    v16h afn = load_afrag(A, mt * 16, lda, ktn * 32, lane);
    v16h bn0 = load_bpack(P, ktn * NTILES + nt, lane);
    v16h bn1 = load_bpack(P, ktn * NTILES + nt + 1, lane);
    a0 = wmma_f16(af, b0, a0);
    a1 = wmma_f16(af, b1, a1);
    af = afn; b0 = bn0; b1 = bn1;
  }
}

__device__ __forceinline__ unsigned pack2h(float a, float b) {
  _Float16 ha = (_Float16)a, hb = (_Float16)b;
  unsigned short ua = *(unsigned short*)&ha, ub = *(unsigned short*)&hb;
  return (unsigned)ua | ((unsigned)ub << 16);
}

// Pack f32 weight (K x N, row-major) into B-fragment order.
__global__ void pack_weights(const float* __restrict__ W, unsigned* __restrict__ dst,
                             int K, int N) {
  int idx = blockIdx.x * 256 + threadIdx.x;
  int total = (K * N) >> 1;
  if (idx >= total) return;
  int i    = idx & 7;
  int lane = (idx >> 3) & 31;
  int tile = idx >> 8;
  int Nt = N >> 4;
  int nt = tile % Nt, kt = tile / Nt;
  int n  = nt * 16 + (lane & 15);
  int kr = kt * 32 + ((lane >> 4) & 1) * 16 + 2 * i;
  dst[idx] = pack2h(W[(size_t)kr * N + n], W[(size_t)(kr + 1) * N + n]);
}

// Expand rel_bias_table (225, 12) -> relb[h][m][n] (12 x 64 x 64 f32)
__global__ void expand_bias(const float* __restrict__ table, float* __restrict__ out) {
  int idx = blockIdx.x * 256 + threadIdx.x;
  if (idx >= 12 * 64 * 64) return;
  int n = idx & 63, m = (idx >> 6) & 63, h = idx >> 12;
  int r0 = (m >> 3) - (n >> 3) + 7;
  int r1 = (m & 7) - (n & 7) + 7;
  out[idx] = table[(r0 * 15 + r1) * 12 + h];
}

// ---------------------------------------------------------------------------
// LDS map (152 KB total, phase-overlaid):
//   [0      , 49152 ) : xn f16 -> q f16          -> y1 f32 low  -> h1 f16
//   [49152  , 98304 ) : x stage f32 low | k f16  -> y1 f32 high -> h1 f16
//   [98304  , 147456) : x stage f32 high | vT    -> y2 f16      -> fc2 stage f32
//   [147456 , 155648) : scratch (LN reductions / per-wave p & O bounce)
// ---------------------------------------------------------------------------
__global__ __launch_bounds__(256) void swin_block_kernel(
    const float* __restrict__ x,
    const float* __restrict__ n1w, const float* __restrict__ n1b,
    const float* __restrict__ qkvb, const float* __restrict__ projb,
    const float* __restrict__ n2w, const float* __restrict__ n2b,
    const float* __restrict__ fc1b, const float* __restrict__ fc2b,
    const unsigned* __restrict__ qkvp, const unsigned* __restrict__ projp,
    const unsigned* __restrict__ fc1p, const unsigned* __restrict__ fc2p,
    const float* __restrict__ relb,
    _Float16* __restrict__ attws,
    float* __restrict__ out) {
  __shared__ __align__(16) char smem[155648];
  _Float16* qbuf = (_Float16*)smem;             // xn, then q
  _Float16* kbuf = (_Float16*)(smem + 49152);   // k
  _Float16* vtb  = (_Float16*)(smem + 98304);   // vT [head][hd][token]
  const float* xs = (const float*)(smem + 49152); // staged x window [c][t] f32
  float*    y1   = (float*)smem;                // proj output f32
  _Float16* y2   = (_Float16*)(smem + 98304);   // LN2 output
  _Float16* h1   = (_Float16*)smem;             // fc1 output
  float* staged  = (float*)(smem + 98304);      // fc2 half-output stage [c][33]
  char*  scr  = smem + 147456;
  float* redA = (float*)scr;
  float* redB = (float*)(scr + 1024);
  float* stats = (float*)(scr + 2048);

  const int g    = blockIdx.x;
  const int bI   = g >> 8;          // batch
  const int hrow = (g >> 1) & 127;  // input row (flat partition!)
  const int wseg = g & 1;           // which 64-pixel segment of the row
  const int a8   = ((g >> 4) & 15) * 8;  // output 2D-merge coords
  const int d8   = (g & 15) * 8;

  const int tid  = threadIdx.x;
  const int lane = tid & 31;
  const int wave = tid >> 5;
  const int nlow = lane & 15;
  const int hf   = (lane >> 4) & 1;

  const size_t xwin = (size_t)bI * 384 * 16384 + (size_t)hrow * 128 + (size_t)wseg * 64;
  _Float16* attg = attws + (size_t)g * 24576;

  // warm the packed QKV weights toward this WGP (global_prefetch_b8)
  for (unsigned off = (unsigned)tid * 64u; off < 221184u; off += 16384u)
    __builtin_prefetch(qkvp + off, 0, 1);

  // ---- stage x window into LDS via async copy (CDNA5 ASYNC path) ----------
  {
    unsigned xs_lds = (unsigned)(size_t)(__attribute__((address_space(3))) char*)(smem + 49152);
    for (int id = tid; id < 6144; id += 256) {
      int c = id >> 4, t4 = (id & 15) * 4;
      const float* gp = x + xwin + (size_t)c * 16384 + t4;
      unsigned lp = xs_lds + (unsigned)id * 16u;
      asm volatile("global_load_async_to_lds_b128 %0, %1, off"
                   :: "v"(lp), "v"(gp) : "memory");
    }
    asm volatile("s_wait_asynccnt 0x0" ::: "memory");
  }
  __syncthreads();

  // ------------------------- LN1 (xs -> xn f16) ----------------------------
  {
    const int t = tid & 63, cg = tid >> 6;
    float s = 0.f, s2 = 0.f;
    for (int c = cg * 96; c < cg * 96 + 96; ++c) {
      float v = xs[c * 64 + t];
      s += v; s2 += v * v;
    }
    redA[cg * 64 + t] = s;
    redB[cg * 64 + t] = s2;
    __syncthreads();
    if (tid < 64) {
      float su = redA[tid] + redA[64 + tid] + redA[128 + tid] + redA[192 + tid];
      float sq = redB[tid] + redB[64 + tid] + redB[128 + tid] + redB[192 + tid];
      float mean = su * (1.f / 384.f);
      float var  = sq * (1.f / 384.f) - mean * mean;
      stats[2 * tid]     = mean;
      stats[2 * tid + 1] = rsqrtf(var + 1e-5f);
    }
    __syncthreads();
    float mean = stats[2 * t], rs = stats[2 * t + 1];
    for (int c = cg * 96; c < cg * 96 + 96; ++c) {
      float v = xs[c * 64 + t];
      qbuf[t * 384 + c] = (_Float16)((v - mean) * rs * n1w[c] + n1b[c]);
    }
  }
  __syncthreads();

  // ---------- preload QKV A-fragments to registers (frees xn for q) --------
  const int mtw = wave & 3;
  const int nh  = wave >> 2;
  v16h afr[12];
#pragma unroll
  for (int kt = 0; kt < 12; ++kt)
    afr[kt] = load_afrag(qbuf, mtw * 16, 384, kt * 32, lane);
  __syncthreads();

  // ------------------------- QKV GEMM (64x384 @ 384x1152) ------------------
  for (int nt = nh * 36; nt < nh * 36 + 36; nt += 2) {
    v8f a0 = {}; v8f a1 = {};
    v16h b0 = load_bpack(qkvp, nt, lane);
    v16h b1 = load_bpack(qkvp, nt + 1, lane);
#pragma unroll
    for (int kt = 0; kt < 12; ++kt) {
      int ktn = (kt + 1 < 12) ? kt + 1 : kt;
      v16h bn0 = load_bpack(qkvp, ktn * 72 + nt, lane);
      v16h bn1 = load_bpack(qkvp, ktn * 72 + nt + 1, lane);
      a0 = wmma_f16(afr[kt], b0, a0);
      a1 = wmma_f16(afr[kt], b1, a1);
      b0 = bn0; b1 = bn1;
    }
#pragma unroll
    for (int e = 0; e < 2; ++e) {
      v8f acc = e ? a1 : a0;
      int ntc = nt + e;
      int col = ntc * 16 + nlow;
      float bs = qkvb[col];
      if (ntc < 24) {                 // q (scaled by hd^-0.5)
#pragma unroll
        for (int r = 0; r < 8; ++r) {
          int m = mtw * 16 + hf * 8 + r;
          qbuf[m * 384 + col] = (_Float16)((acc[r] + bs) * 0.17677669529663687f);
        }
      } else if (ntc < 48) {          // k
        int c2 = col - 384;
#pragma unroll
        for (int r = 0; r < 8; ++r) {
          int m = mtw * 16 + hf * 8 + r;
          kbuf[m * 384 + c2] = (_Float16)(acc[r] + bs);
        }
      } else {                        // v, stored transposed per head
        int c2 = col - 768;
        int hh = c2 >> 5, hd = c2 & 31;
#pragma unroll
        for (int r = 0; r < 8; ++r) {
          int m = mtw * 16 + hf * 8 + r;
          vtb[hh * 2048 + hd * 64 + m] = (_Float16)(acc[r] + bs);
        }
      }
    }
  }
  __syncthreads();

  // -------------- attention: 12 heads x 4 row-tiles = 48 items -------------
  for (int it = wave; it < 48; it += 8) {
    int h = it >> 2, mt = it & 3;
    _Float16* pscr = (_Float16*)scr + wave * 512;  // 1KB per-wave bounce
    v16h aq = load_afrag(qbuf, mt * 16, 384, h * 32, lane);
    v8f S[4];
#pragma unroll
    for (int nt = 0; nt < 4; ++nt) {
      v16h bk = load_bfrag(kbuf + (size_t)(nt * 16 + nlow) * 384 + h * 32 + hf * 16);
      v8f z = {};
      S[nt] = wmma_f16(aq, bk, z);
    }
    // relative-position bias
#pragma unroll
    for (int nt = 0; nt < 4; ++nt)
#pragma unroll
      for (int r = 0; r < 8; ++r) {
        int m = mt * 16 + hf * 8 + r;
        S[nt][r] += relb[h * 4096 + m * 64 + nt * 16 + nlow];
      }
    // softmax over 64 keys (rows live in a fixed 16-lane half)
#pragma unroll
    for (int r = 0; r < 8; ++r) {
      float mx = fmaxf(fmaxf(S[0][r], S[1][r]), fmaxf(S[2][r], S[3][r]));
      mx = fmaxf(mx, __shfl_xor(mx, 1));
      mx = fmaxf(mx, __shfl_xor(mx, 2));
      mx = fmaxf(mx, __shfl_xor(mx, 4));
      mx = fmaxf(mx, __shfl_xor(mx, 8));
      float e0 = __expf(S[0][r] - mx);
      float e1 = __expf(S[1][r] - mx);
      float e2 = __expf(S[2][r] - mx);
      float e3 = __expf(S[3][r] - mx);
      float sm = e0 + e1 + e2 + e3;
      sm += __shfl_xor(sm, 1);
      sm += __shfl_xor(sm, 2);
      sm += __shfl_xor(sm, 4);
      sm += __shfl_xor(sm, 8);
      float inv = 1.f / sm;
      S[0][r] = e0 * inv; S[1][r] = e1 * inv; S[2][r] = e2 * inv; S[3][r] = e3 * inv;
    }
    // p @ v : bounce p D-layout -> A-layout through per-wave LDS scratch
    v8f O0 = {}; v8f O1 = {};
#pragma unroll
    for (int kt = 0; kt < 2; ++kt) {
#pragma unroll
      for (int nn = 0; nn < 2; ++nn)
#pragma unroll
        for (int r = 0; r < 8; ++r)
          pscr[(hf * 8 + r) * 32 + nn * 16 + nlow] = (_Float16)S[kt * 2 + nn][r];
      v16h ap  = load_afrag(pscr, 0, 32, 0, lane);
      v16h bv0 = load_bfrag(vtb + h * 2048 + (size_t)nlow * 64 + kt * 32 + hf * 16);
      v16h bv1 = load_bfrag(vtb + h * 2048 + (size_t)(16 + nlow) * 64 + kt * 32 + hf * 16);
      O0 = wmma_f16(ap, bv0, O0);
      O1 = wmma_f16(ap, bv1, O1);
    }
    // bounce O tile to LDS, then two coalesced b128 global stores per lane
#pragma unroll
    for (int r = 0; r < 8; ++r) {
      pscr[(hf * 8 + r) * 32 + nlow]      = (_Float16)O0[r];
      pscr[(hf * 8 + r) * 32 + 16 + nlow] = (_Float16)O1[r];
    }
    {
      const u32x4* ps = (const u32x4*)pscr;  // 16 rows x 4 uint4
#pragma unroll
      for (int j = 0; j < 2; ++j) {
        int id = lane + j * 32;
        int row = id >> 2, qq = id & 3;
        u32x4 vq = ps[row * 4 + qq];
        *(u32x4*)(attg + (size_t)(mt * 16 + row) * 384 + h * 32 + qq * 8) = vq;
      }
    }
  }
  __syncthreads();

  // ------------------------- proj GEMM (attn_out @ proj_w) -----------------
  for (int tp = wave; tp < 48; tp += 8) {
    int mt = tp / 12, nt = (tp % 12) * 2;
    v8f a0 = {}; v8f a1 = {};
    gemm_pair<12, 24>(attg, 384, mt, projp, nt, lane, a0, a1);
#pragma unroll
    for (int e = 0; e < 2; ++e) {
      v8f acc = e ? a1 : a0;
      int col = (nt + e) * 16 + nlow;
      float bs = projb[col];
#pragma unroll
      for (int r = 0; r < 8; ++r) {
        int m = mt * 16 + hf * 8 + r;
        y1[m * 384 + col] = acc[r] + bs;
      }
    }
  }
  __syncthreads();

  // ------------------------- LN2 (y1 f32 -> y2 f16) ------------------------
  {
    const int t = tid & 63, cg = tid >> 6;
    float s = 0.f, s2 = 0.f;
    for (int c = cg * 96; c < cg * 96 + 96; ++c) {
      float v = y1[t * 384 + c];
      s += v; s2 += v * v;
    }
    redA[cg * 64 + t] = s;
    redB[cg * 64 + t] = s2;
    __syncthreads();
    if (tid < 64) {
      float su = redA[tid] + redA[64 + tid] + redA[128 + tid] + redA[192 + tid];
      float sq = redB[tid] + redB[64 + tid] + redB[128 + tid] + redB[192 + tid];
      float mean = su * (1.f / 384.f);
      float var  = sq * (1.f / 384.f) - mean * mean;
      stats[2 * tid]     = mean;
      stats[2 * tid + 1] = rsqrtf(var + 1e-5f);
    }
    __syncthreads();
    float mean = stats[2 * t], rs = stats[2 * t + 1];
    for (int c = cg * 96; c < cg * 96 + 96; ++c) {
      float v = y1[t * 384 + c];
      y2[t * 384 + c] = (_Float16)((v - mean) * rs * n2w[c] + n2b[c]);
    }
  }
  __syncthreads();

  // ------------------------- fc1 + exact GELU ------------------------------
  for (int tp = wave; tp < 96; tp += 8) {
    int mt = tp / 24, nt = (tp % 24) * 2;
    v8f a0 = {}; v8f a1 = {};
    gemm_pair<12, 48>(y2, 384, mt, fc1p, nt, lane, a0, a1);
#pragma unroll
    for (int e = 0; e < 2; ++e) {
      v8f acc = e ? a1 : a0;
      int col = (nt + e) * 16 + nlow;
      float bs = fc1b[col];
#pragma unroll
      for (int r = 0; r < 8; ++r) {
        int m = mt * 16 + hf * 8 + r;
        float v = acc[r] + bs;
        float ge = 0.5f * v * (1.f + erff(v * 0.70710678118654752f));
        h1[m * 768 + col] = (_Float16)ge;
      }
    }
  }
  __syncthreads();

  // ---------- fc2 + 2D window-merge + residual, in two 32-row halves -------
  // Stage each half in LDS ([c][33] padded, conflict-free) so output stores
  // and residual x loads are 32B-contiguous per 8-lane group instead of 4B
  // scatter: this is the dominant HBM traffic of the whole kernel.
  for (int mh = 0; mh < 2; ++mh) {
    for (int tp = wave; tp < 24; tp += 8) {
      int mt = mh * 2 + tp / 12, nt = (tp % 12) * 2;
      v8f a0 = {}; v8f a1 = {};
      gemm_pair<24, 24>(h1, 768, mt, fc2p, nt, lane, a0, a1);
#pragma unroll
      for (int e = 0; e < 2; ++e) {
        v8f acc = e ? a1 : a0;
        int col = (nt + e) * 16 + nlow;
        float bs = fc2b[col];
#pragma unroll
        for (int r = 0; r < 8; ++r) {
          int m = mt * 16 + hf * 8 + r;
          staged[col * 33 + (m - mh * 32)] = acc[r] + bs;
        }
      }
    }
    __syncthreads();
    for (int item = tid; item < 12288; item += 256) {
      int c = item >> 5, tl = item & 31;
      int t = mh * 32 + tl;
      int ho = a8 + (t >> 3), wo = d8 + (t & 7);
      size_t addr = (size_t)(bI * 384 + c) * 16384 + (size_t)ho * 128 + wo;
      out[addr] = staged[c * 33 + tl] + x[addr];
    }
    __syncthreads();
  }
}

// ---------------------------------------------------------------------------
// Workspace layout (bytes):
static const size_t OFF_QKVP = 0;         // 884736
static const size_t OFF_PROJP = 884736;   // 294912
static const size_t OFF_FC1P = 1179648;   // 589824
static const size_t OFF_FC2P = 1769472;   // 589824
static const size_t OFF_RELB = 2359296;   // 196608
static const size_t OFF_ATT  = 2555904;   // 100663296 (2048 x 64 x 384 f16)
// total ~103 MB

extern "C" void kernel_launch(void* const* d_in, const int* in_sizes, int n_in,
                              void* d_out, int out_size, void* d_ws, size_t ws_size,
                              hipStream_t stream) {
  (void)in_sizes; (void)n_in; (void)out_size; (void)ws_size;
  const float* x      = (const float*)d_in[0];
  const float* n1w    = (const float*)d_in[3];
  const float* n1b    = (const float*)d_in[4];
  const float* qkv_w  = (const float*)d_in[5];
  const float* qkv_b  = (const float*)d_in[6];
  const float* proj_w = (const float*)d_in[7];
  const float* proj_b = (const float*)d_in[8];
  const float* rtab   = (const float*)d_in[9];
  const float* n2w    = (const float*)d_in[10];
  const float* n2b    = (const float*)d_in[11];
  const float* fc1_w  = (const float*)d_in[12];
  const float* fc1_b  = (const float*)d_in[13];
  const float* fc2_w  = (const float*)d_in[14];
  const float* fc2_b  = (const float*)d_in[15];

  unsigned char* ws = (unsigned char*)d_ws;
  unsigned* qkvp = (unsigned*)(ws + OFF_QKVP);
  unsigned* projp = (unsigned*)(ws + OFF_PROJP);
  unsigned* fc1p = (unsigned*)(ws + OFF_FC1P);
  unsigned* fc2p = (unsigned*)(ws + OFF_FC2P);
  float* relb = (float*)(ws + OFF_RELB);
  _Float16* attg = (_Float16*)(ws + OFF_ATT);

  pack_weights<<<864, 256, 0, stream>>>(qkv_w, qkvp, 384, 1152);
  pack_weights<<<288, 256, 0, stream>>>(proj_w, projp, 384, 384);
  pack_weights<<<576, 256, 0, stream>>>(fc1_w, fc1p, 384, 768);
  pack_weights<<<576, 256, 0, stream>>>(fc2_w, fc2p, 768, 384);
  expand_bias<<<192, 256, 0, stream>>>(rtab, relb);

  swin_block_kernel<<<2048, 256, 0, stream>>>(
      x, n1w, n1b, qkv_b, proj_b, n2w, n2b, fc1_b, fc2_b,
      qkvp, projp, fc1p, fc2p, relb, attg, (float*)d_out);
}
